// RNNLanguageModel_74019466379625
// MI455X (gfx1250) — matrix operand order repaired and checked by reading, not compiled, not run on hardware
//
#include <hip/hip_runtime.h>

#define BATCH 16
#define SEQ   128
#define EMBED 256
#define UNITS 1024
#define G4    (4*UNITS)     // 4096
#define VOCAB 32000
#define ROWS  (BATCH*SEQ)   // 2048

typedef __attribute__((ext_vector_type(16))) __bf16 v16bf;
typedef __attribute__((ext_vector_type(8)))  float  v8f;

union Frag16 {
    v16bf v;
    uint4 u[2];
};

#define WMMA_BF16(A, B, C) \
    __builtin_amdgcn_wmma_f32_16x16x32_bf16(false, (A).v, false, (B).v, (short)0, (C), false, false)

__device__ __forceinline__ unsigned short f2bf(float f) {
    unsigned int u = __float_as_uint(f);
    u += 0x7FFFu + ((u >> 16) & 1u);      // round-to-nearest-even
    return (unsigned short)(u >> 16);
}

__device__ __forceinline__ float sigm(float x) { return 1.0f / (1.0f + expf(-x)); }

__device__ __forceinline__ void loadA(Frag16& f, const unsigned short* p) {
    f.u[0] = *(const uint4*)(p);
    f.u[1] = *(const uint4*)(p + 16);
}
__device__ __forceinline__ void loadB(Frag16& f, const unsigned short* p) {
    f.u[0] = *(const uint4*)(p);
    f.u[1] = *(const uint4*)(p + 8);
}

// ---------------------------------------------------------------------------
// Embedding gather + fp32 -> bf16 convert: Xbf[r][e] = bf16(emb[tok[r]][e])
// ---------------------------------------------------------------------------
__global__ __launch_bounds__(EMBED) void embed_gather(
    const int* __restrict__ tok, const float* __restrict__ emb,
    unsigned short* __restrict__ Xbf)
{
    int r = blockIdx.x;          // 0..2047, r = b*SEQ + t (inputs row-major [B][T])
    int e = threadIdx.x;         // 0..255
    int tk = tok[r];
    Xbf[(size_t)r * EMBED + e] = f2bf(emb[(size_t)tk * EMBED + e]);
}

// ---------------------------------------------------------------------------
// Swizzle row-major fp32 B[K][N] into bf16 WMMA B-fragment blocks:
//   block (nt,kt) of 512 bf16 at ((nt*KT + kt) * 512); lane holds 16 contiguous
//   values: lane n (0..15): K = kt*32 + (idx&7) + 16*(idx>>3), col = nt*16 + n
//           lane n+16     : K = kt*32 + 8 + (idx&7) + 16*(idx>>3)
// ---------------------------------------------------------------------------
__global__ __launch_bounds__(256) void swizzleB(
    const float* __restrict__ in, unsigned short* __restrict__ out, int K, int N)
{
    size_t tid = (size_t)blockIdx.x * 256 + threadIdx.x;
    size_t total = (size_t)K * N;
    if (tid >= total) return;
    int KT  = K >> 5;
    size_t blk = tid >> 9;
    int rem  = (int)(tid & 511);
    int lane = rem >> 4;
    int idx  = rem & 15;
    int kt = (int)(blk % KT);
    int nt = (int)(blk / KT);
    int row = kt * 32 + ((lane >> 4) << 3) + (idx & 7) + ((idx >> 3) << 4);
    int col = nt * 16 + (lane & 15);
    out[tid] = f2bf(in[(size_t)row * N + col]);
}

// ---------------------------------------------------------------------------
// Zero LSTM state
// ---------------------------------------------------------------------------
__global__ __launch_bounds__(256) void init_state(
    float* __restrict__ c, unsigned short* __restrict__ h0, unsigned short* __restrict__ h1)
{
    int i = blockIdx.x * 256 + threadIdx.x;
    if (i < BATCH * UNITS) { c[i] = 0.0f; h0[i] = 0; h1[i] = 0; }
}

// ---------------------------------------------------------------------------
// Generic bf16 GEMM: C[M][N] = A[M][K](bf16 rowmajor) * Bswz + bias
// block = 128 threads = 4 waves; each wave computes a 32x64 tile
// (2 M-tiles x 4 N-tiles, 8 WMMAs per k-step, register double-buffered).
// grid = (N/64, M/128)
// ---------------------------------------------------------------------------
__global__ __launch_bounds__(128) void gemm_bf16(
    const unsigned short* __restrict__ A,
    const unsigned short* __restrict__ Bs,
    const float* __restrict__ bias,
    float* __restrict__ C,
    int M, int N, int K)
{
    const int lane   = threadIdx.x & 31;
    const int wave   = threadIdx.x >> 5;
    const int KT     = K >> 5;
    const int mpair  = blockIdx.y * 4 + wave;   // rows [mpair*32, mpair*32+32)
    const int ntile0 = blockIdx.x * 4;

    const int row0   = mpair * 32 + (lane & 15);
    const int k0sel  = (lane >> 4) << 3;        // 0 or 8
    const unsigned short* pa0 = A + (size_t)row0 * K + k0sel;
    const unsigned short* pa1 = pa0 + (size_t)16 * K;
    const unsigned short* pb  = Bs + (((size_t)ntile0 * KT) << 9) + (lane << 4);
    const size_t nstep = (size_t)KT << 9;

    v8f acc00 = {}, acc01 = {}, acc02 = {}, acc03 = {};
    v8f acc10 = {}, acc11 = {}, acc12 = {}, acc13 = {};

    Frag16 a0c, a1c, b0c, b1c, b2c, b3c;
    loadA(a0c, pa0);
    loadA(a1c, pa1);
    loadB(b0c, pb);
    loadB(b1c, pb + nstep);
    loadB(b2c, pb + 2 * nstep);
    loadB(b3c, pb + 3 * nstep);

    for (int kt = 0; kt < KT - 1; ++kt) {
        // prefetch next k-step into shadow registers (independent of WMMAs)
        Frag16 a0n, a1n, b0n, b1n, b2n, b3n;
        const unsigned short* qa = pa0 + (kt + 1) * 32;
        const unsigned short* qb = pb + ((size_t)(kt + 1) << 9);
        loadA(a0n, qa);
        loadA(a1n, qa + (size_t)16 * K);
        loadB(b0n, qb);
        loadB(b1n, qb + nstep);
        loadB(b2n, qb + 2 * nstep);
        loadB(b3n, qb + 3 * nstep);

        acc00 = WMMA_BF16(a0c, b0c, acc00);
        acc01 = WMMA_BF16(a0c, b1c, acc01);
        acc02 = WMMA_BF16(a0c, b2c, acc02);
        acc03 = WMMA_BF16(a0c, b3c, acc03);
        acc10 = WMMA_BF16(a1c, b0c, acc10);
        acc11 = WMMA_BF16(a1c, b1c, acc11);
        acc12 = WMMA_BF16(a1c, b2c, acc12);
        acc13 = WMMA_BF16(a1c, b3c, acc13);

        a0c = a0n; a1c = a1n;
        b0c = b0n; b1c = b1n; b2c = b2n; b3c = b3n;
    }
    acc00 = WMMA_BF16(a0c, b0c, acc00);
    acc01 = WMMA_BF16(a0c, b1c, acc01);
    acc02 = WMMA_BF16(a0c, b2c, acc02);
    acc03 = WMMA_BF16(a0c, b3c, acc03);
    acc10 = WMMA_BF16(a1c, b0c, acc10);
    acc11 = WMMA_BF16(a1c, b1c, acc11);
    acc12 = WMMA_BF16(a1c, b2c, acc12);
    acc13 = WMMA_BF16(a1c, b3c, acc13);

    const int srow = mpair * 32 + ((lane >> 4) << 3);
    const int c0   = ntile0 * 16 + (lane & 15);
    const float bv0 = bias[c0];
    const float bv1 = bias[c0 + 16];
    const float bv2 = bias[c0 + 32];
    const float bv3 = bias[c0 + 48];
    #pragma unroll
    for (int r = 0; r < 8; ++r) {
        float* cp0 = C + (size_t)(srow + r) * N;
        float* cp1 = C + (size_t)(srow + 16 + r) * N;
        cp0[c0     ] = acc00[r] + bv0;
        cp0[c0 + 16] = acc01[r] + bv1;
        cp0[c0 + 32] = acc02[r] + bv2;
        cp0[c0 + 48] = acc03[r] + bv3;
        cp1[c0     ] = acc10[r] + bv0;
        cp1[c0 + 16] = acc11[r] + bv1;
        cp1[c0 + 32] = acc12[r] + bv2;
        cp1[c0 + 48] = acc13[r] + bv3;
    }
}

// ---------------------------------------------------------------------------
// One LSTM timestep: z = Zx[t] + h@U ; gates ; update c, h  (bias folded in Zx)
// 16 blocks x 128 threads; block owns 64 hidden units across all 4 gates.
// wave w computes gate w's 64 columns (4 WMMA N-tiles), K=1024, pipelined.
// ---------------------------------------------------------------------------
__global__ __launch_bounds__(128) void lstm_step(
    const unsigned short* __restrict__ hin,    // [16][1024] bf16
    const unsigned short* __restrict__ Uswz,
    const float* __restrict__ Zx,              // [2048][4096] fp32, row = b*SEQ + t
    float* __restrict__ cst,                   // [16][1024] fp32
    unsigned short* __restrict__ hout,         // [16][1024] bf16
    unsigned short* __restrict__ hsbf,         // [2048][1024] bf16
    int t)
{
    __shared__ float zs[4][BATCH][64];

    const int lane = threadIdx.x & 31;
    const int w    = threadIdx.x >> 5;          // gate: 0=i 1=f 2=g 3=o
    const int hb   = blockIdx.x * 64;           // hidden base for this block
    const int KT   = UNITS >> 5;                // 32
    const int ntile0 = (w * UNITS + hb) >> 4;

    const int arow  = lane & 15;                // batch index
    const int k0sel = (lane >> 4) << 3;
    const unsigned short* pa = hin + (size_t)arow * UNITS + k0sel;
    const unsigned short* pb = Uswz + (((size_t)ntile0 * KT) << 9) + (lane << 4);
    const size_t nstep = (size_t)KT << 9;

    v8f acc0 = {}, acc1 = {}, acc2 = {}, acc3 = {};

    Frag16 ac, b0c, b1c, b2c, b3c;
    loadA(ac, pa);
    loadB(b0c, pb);
    loadB(b1c, pb + nstep);
    loadB(b2c, pb + 2 * nstep);
    loadB(b3c, pb + 3 * nstep);

    for (int kt = 0; kt < KT - 1; ++kt) {
        Frag16 an, b0n, b1n, b2n, b3n;
        const unsigned short* qb = pb + ((size_t)(kt + 1) << 9);
        loadA(an, pa + (kt + 1) * 32);
        loadB(b0n, qb);
        loadB(b1n, qb + nstep);
        loadB(b2n, qb + 2 * nstep);
        loadB(b3n, qb + 3 * nstep);

        acc0 = WMMA_BF16(ac, b0c, acc0);
        acc1 = WMMA_BF16(ac, b1c, acc1);
        acc2 = WMMA_BF16(ac, b2c, acc2);
        acc3 = WMMA_BF16(ac, b3c, acc3);

        ac = an; b0c = b0n; b1c = b1n; b2c = b2n; b3c = b3n;
    }
    acc0 = WMMA_BF16(ac, b0c, acc0);
    acc1 = WMMA_BF16(ac, b1c, acc1);
    acc2 = WMMA_BF16(ac, b2c, acc2);
    acc3 = WMMA_BF16(ac, b3c, acc3);

    // scatter z = acc + Zx into LDS (bias b already folded into Zx)
    const int mofs = (lane >> 4) << 3;
    const int lc   = lane & 15;
    const int gbase = w * UNITS + hb;
    #pragma unroll
    for (int r = 0; r < 8; ++r) {
        int bb = mofs + r;
        const float* zrow = Zx + ((size_t)bb * SEQ + t) * G4 + gbase;
        zs[w][bb][lc     ] = acc0[r] + zrow[lc     ];
        zs[w][bb][lc + 16] = acc1[r] + zrow[lc + 16];
        zs[w][bb][lc + 32] = acc2[r] + zrow[lc + 32];
        zs[w][bb][lc + 48] = acc3[r] + zrow[lc + 48];
    }
    __syncthreads();

    // elementwise gate math: 128 threads x 8 = 16 batches x 64 hiddens
    int bb = threadIdx.x >> 3;
    int j0 = (threadIdx.x & 7) * 8;
    #pragma unroll
    for (int s = 0; s < 8; ++s) {
        int j = j0 + s;
        int hidx = hb + j;
        float iv = zs[0][bb][j];
        float fv = zs[1][bb][j];
        float gv = zs[2][bb][j];
        float ov = zs[3][bb][j];
        float cold = cst[bb * UNITS + hidx];
        float cn = sigm(fv) * cold + sigm(iv) * tanhf(gv);
        cst[bb * UNITS + hidx] = cn;
        unsigned short hv = f2bf(sigm(ov) * tanhf(cn));
        hout[bb * UNITS + hidx] = hv;
        hsbf[((size_t)bb * SEQ + t) * UNITS + hidx] = hv;
    }
}

// ---------------------------------------------------------------------------
extern "C" void kernel_launch(void* const* d_in, const int* in_sizes, int n_in,
                              void* d_out, int out_size, void* d_ws, size_t ws_size,
                              hipStream_t stream) {
    const int*   tok = (const int*)d_in[0];
    const float* emb = (const float*)d_in[1];
    const float* W   = (const float*)d_in[2];
    const float* U   = (const float*)d_in[3];
    const float* b   = (const float*)d_in[4];
    const float* Wc  = (const float*)d_in[5];
    const float* bc  = (const float*)d_in[6];
    float* out = (float*)d_out;

    char* p = (char*)d_ws;
    auto carve = [&](size_t bytes) -> char* {
        char* r = p;
        p += (bytes + 255) & ~(size_t)255;
        return r;
    };
    unsigned short* Xbf   = (unsigned short*)carve((size_t)ROWS  * EMBED * 2);
    unsigned short* Wswz  = (unsigned short*)carve((size_t)EMBED * G4    * 2);
    unsigned short* Uswz  = (unsigned short*)carve((size_t)UNITS * G4    * 2);
    unsigned short* Wcswz = (unsigned short*)carve((size_t)UNITS * VOCAB * 2);
    float*          Zx    = (float*)         carve((size_t)ROWS  * G4    * 4);
    float*          cst   = (float*)         carve((size_t)BATCH * UNITS * 4);
    unsigned short* hA    = (unsigned short*)carve((size_t)BATCH * UNITS * 2);
    unsigned short* hB    = (unsigned short*)carve((size_t)BATCH * UNITS * 2);
    unsigned short* hsbf  = (unsigned short*)carve((size_t)ROWS  * UNITS * 2);

    // Prep: embedding gather, weight conversion/swizzle
    hipLaunchKernelGGL(embed_gather, dim3(ROWS), dim3(EMBED), 0, stream, tok, emb, Xbf);
    {
        size_t tot = (size_t)EMBED * G4;
        hipLaunchKernelGGL(swizzleB, dim3((unsigned)((tot + 255) / 256)), dim3(256), 0, stream, W, Wswz, EMBED, G4);
    }
    {
        size_t tot = (size_t)UNITS * G4;
        hipLaunchKernelGGL(swizzleB, dim3((unsigned)((tot + 255) / 256)), dim3(256), 0, stream, U, Uswz, UNITS, G4);
    }
    {
        size_t tot = (size_t)UNITS * VOCAB;
        hipLaunchKernelGGL(swizzleB, dim3((unsigned)((tot + 255) / 256)), dim3(256), 0, stream, Wc, Wcswz, UNITS, VOCAB);
    }

    // Phase 1: Zx = X @ W + b   (bias b folded here, once)
    hipLaunchKernelGGL(gemm_bf16, dim3(G4 / 64, ROWS / 128), dim3(128), 0, stream,
                       Xbf, Wswz, b, Zx, ROWS, G4, EMBED);

    // Phase 2: recurrence over SEQ timesteps, h double-buffered
    hipLaunchKernelGGL(init_state, dim3((BATCH * UNITS + 255) / 256), dim3(256), 0, stream, cst, hA, hB);
    for (int t = 0; t < SEQ; ++t) {
        const unsigned short* hin = (t & 1) ? hB : hA;
        unsigned short*      hout = (t & 1) ? hA : hB;
        hipLaunchKernelGGL(lstm_step, dim3(UNITS / 64), dim3(128), 0, stream,
                           hin, Uswz, Zx, cst, hout, hsbf, t);
    }

    // Phase 3: logits = hs @ Wc + bc
    hipLaunchKernelGGL(gemm_bf16, dim3(VOCAB / 64, ROWS / 128), dim3(128), 0, stream,
                       hsbf, Wcswz, bc, out, ROWS, VOCAB, UNITS);
}
